// LambdaLayer1D_new_80556406604343
// MI455X (gfx1250) — matrix operand order repaired and checked by reading, not compiled, not run on hardware
//
#include <hip/hip_runtime.h>
#include <hip/hip_bf16.h>

// ---------------- problem constants ----------------
#define Bc     16
#define INTOK  1024
#define DIMc   1024
#define HEADS  4
#define KKc    16
#define VVc    256
#define MMc    23
#define PADc   11
#define OTOT   336            // 64 (q) + 16 (k) + 256 (v) fused projection rows
#define EPSc   1e-5f

typedef __attribute__((ext_vector_type(16))) _Float16 v16h;
typedef __attribute__((ext_vector_type(8)))  _Float16 v8h;
typedef __attribute__((ext_vector_type(8)))  float    v8f;
typedef __attribute__((ext_vector_type(2)))  float    v2f;

#define LDS_STRIDE 72   // 64 + 8 pad: keeps every fragment half 16B-aligned

#define CAT16(lo, hi) __builtin_shufflevector(lo, hi, 0,1,2,3,4,5,6,7,8,9,10,11,12,13,14,15)

// =====================================================================
// K1: fused QKV projection GEMM via v_wmma_f32_16x16x32_f16
//     P[b][o][d] = sum_i W[o][i] * x[b][i][d] + bias[o]
// 64x64 tile per block (4 waves, 32x32 each), K staged 64 at a time.
// Fragments assembled with ds_load_b128 (each half-fragment is contiguous).
// =====================================================================
__global__ __launch_bounds__(128)
void k_proj_gemm(const float* __restrict__ x,
                 const float* __restrict__ Wq, const float* __restrict__ bq,
                 const float* __restrict__ Wk, const float* __restrict__ bk,
                 const float* __restrict__ Wv, const float* __restrict__ bv,
                 float* __restrict__ P)
{
  __shared__ __align__(16) _Float16 Ah[64 * LDS_STRIDE];  // W tile [o_local][i_local]
  __shared__ __align__(16) _Float16 Bt[64 * LDS_STRIDE];  // x tile transposed [d_local][i_local]

  const int b   = blockIdx.z;
  const int m0  = blockIdx.y * 64;          // o tile base
  const int n0  = blockIdx.x * 64;          // d tile base
  const int tid = threadIdx.x;
  const int w    = tid >> 5, lane = tid & 31;
  const int m_off = (w & 1) * 32, n_off = (w >> 1) * 32;
  const int Ml = lane & 15, g = lane >> 4;

  const float* xb = x + (size_t)b * INTOK * DIMc;

  v8f acc[2][2] = {};

  for (int k0 = 0; k0 < INTOK; k0 += 64) {
    // ---- stage tiles (float4 global loads, f32->f16 into LDS) ----
    for (int idx = tid; idx < 1024; idx += 128) {
      int r = idx >> 4, c = (idx & 15) << 2;
      int o = m0 + r;
      float4 av = make_float4(0.f, 0.f, 0.f, 0.f);
      if (o < OTOT) {
        const float* wr = (o < 64) ? (Wq + (size_t)o * INTOK)
                        : (o < 80) ? (Wk + (size_t)(o - 64) * INTOK)
                                   : (Wv + (size_t)(o - 80) * INTOK);
        av = *(const float4*)(wr + k0 + c);
        if (k0 + 64 < INTOK) __builtin_prefetch(wr + k0 + 64 + c, 0, 0);
      }
      _Float16* ap = &Ah[r * LDS_STRIDE + c];
      ap[0] = (_Float16)av.x; ap[1] = (_Float16)av.y;
      ap[2] = (_Float16)av.z; ap[3] = (_Float16)av.w;

      float4 bv4 = *(const float4*)(xb + (size_t)(k0 + r) * DIMc + n0 + c);
      if (k0 + 64 < INTOK) __builtin_prefetch(xb + (size_t)(k0 + 64 + r) * DIMc + n0 + c, 0, 0);
      Bt[(c + 0) * LDS_STRIDE + r] = (_Float16)bv4.x;
      Bt[(c + 1) * LDS_STRIDE + r] = (_Float16)bv4.y;
      Bt[(c + 2) * LDS_STRIDE + r] = (_Float16)bv4.z;
      Bt[(c + 3) * LDS_STRIDE + r] = (_Float16)bv4.w;
    }
    __syncthreads();

    #pragma unroll
    for (int kh = 0; kh < 64; kh += 32) {
      v16h afr[2], bfr[2];
      #pragma unroll
      for (int s = 0; s < 2; ++s) {
        // A 16x32 f16: elems 0..7 = K[8g..8g+7], elems 8..15 = K[16+8g..+7]  (contiguous)
        const _Float16* arow = &Ah[(m_off + s * 16 + Ml) * LDS_STRIDE + kh];
        v8h alo = *(const v8h*)(arow + 8 * g);
        v8h ahi = *(const v8h*)(arow + 16 + 8 * g);
        afr[s] = CAT16(alo, ahi);
        // B 32x16 f16: elems 0..7 = K[8g..8g+7], elems 8..15 = K[16+8g..+7] at column N
        const _Float16* brow = &Bt[(n_off + s * 16 + Ml) * LDS_STRIDE + kh + 8 * g];
        v8h blo = *(const v8h*)(brow);
        v8h bhi = *(const v8h*)(brow + 16);
        bfr[s] = CAT16(blo, bhi);
      }
      #pragma unroll
      for (int sm = 0; sm < 2; ++sm)
        #pragma unroll
        for (int sn = 0; sn < 2; ++sn)
          acc[sm][sn] = __builtin_amdgcn_wmma_f32_16x16x32_f16(
              false, afr[sm], false, bfr[sn], (short)0, acc[sm][sn], false, false);
    }
    __syncthreads();
  }

  // ---- epilogue: bias add + store (C layout: VGPR r -> M = r + 8g, N = lane&15) ----
  #pragma unroll
  for (int sm = 0; sm < 2; ++sm) {
    #pragma unroll
    for (int sn = 0; sn < 2; ++sn) {
      #pragma unroll
      for (int r = 0; r < 8; ++r) {
        int o = m0 + m_off + sm * 16 + r + 8 * g;
        int d = n0 + n_off + sn * 16 + Ml;
        if (o < OTOT) {
          float bias = (o < 64) ? bq[o] : (o < 80) ? bk[o - 64] : bv[o - 80];
          P[((size_t)(b * OTOT + o)) * DIMc + d] = acc[sm][sn][r] + bias;
        }
      }
    }
  }
}

// =====================================================================
// K2: BatchNorm statistics per channel d (over batch x L), q and v regions
// =====================================================================
__global__ __launch_bounds__(256)
void k_bnstats(const float* __restrict__ P, float* __restrict__ stats)
{
  __shared__ float r1[256], r2[256];
  const int d = blockIdx.x, t = threadIdx.x;

  float s1 = 0.f, s2 = 0.f;
  for (int i = t; i < 16 * 64; i += 256) {
    int bb = i >> 6, o = i & 63;
    float v = P[((size_t)(bb * OTOT + o)) * DIMc + d];
    s1 += v; s2 += v * v;
  }
  r1[t] = s1; r2[t] = s2; __syncthreads();
  for (int s = 128; s > 0; s >>= 1) {
    if (t < s) { r1[t] += r1[t + s]; r2[t] += r2[t + s]; }
    __syncthreads();
  }
  if (t == 0) {
    float mean = r1[0] * (1.f / 1024.f);
    float var  = r2[0] * (1.f / 1024.f) - mean * mean;
    stats[d]        = mean;
    stats[1024 + d] = rsqrtf(var + EPSc);
  }
  __syncthreads();

  s1 = 0.f; s2 = 0.f;
  for (int i = t; i < 16 * 256; i += 256) {
    int bb = i >> 8, o = 80 + (i & 255);
    float v = P[((size_t)(bb * OTOT + o)) * DIMc + d];
    s1 += v; s2 += v * v;
  }
  r1[t] = s1; r2[t] = s2; __syncthreads();
  for (int s = 128; s > 0; s >>= 1) {
    if (t < s) { r1[t] += r1[t + s]; r2[t] += r2[t + s]; }
    __syncthreads();
  }
  if (t == 0) {
    float mean = r1[0] * (1.f / 4096.f);
    float var  = r2[0] * (1.f / 4096.f) - mean * mean;
    stats[2048 + d] = mean;
    stats[3072 + d] = rsqrtf(var + EPSc);
  }
}

// =====================================================================
// K3: BN affine + scatter into .view() layouts
//   q[b][h][n][k]  (k innermost -> b128 loads in K6)
//   v[b][vi][n]    f32 + f16 mirror
// =====================================================================
__global__ __launch_bounds__(256)
void k_scatter_qv(const float* __restrict__ P, const float* __restrict__ stats,
                  const float* __restrict__ gq, const float* __restrict__ betq,
                  const float* __restrict__ gv, const float* __restrict__ betv,
                  float* __restrict__ q_arr, float* __restrict__ v_arr,
                  _Float16* __restrict__ v_h)
{
  const int d = blockIdx.x, b = blockIdx.y, t = threadIdx.x;
  const float mq = stats[d],        rq = stats[1024 + d];
  const float mv = stats[2048 + d], rv = stats[3072 + d];
  const float gsq = gq[d] * rq, bsq = betq[d] - mq * gq[d] * rq;
  const float gsv = gv[d] * rv, bsv = betv[d] - mv * gv[d] * rv;

  for (int o = t; o < OTOT; o += 256) {
    if (o < 64) {
      float val = P[((size_t)(b * OTOT + o)) * DIMc + d] * gsq + bsq;
      int h = d >> 8, k = (d >> 4) & 15, n = (d & 15) * 64 + o;
      q_arr[(((size_t)(b * 4 + h)) * 1024 + n) * 16 + k] = val;
    } else if (o >= 80) {
      int ov = o - 80;
      float val = P[((size_t)(b * OTOT + o)) * DIMc + d] * gsv + bsv;
      int vi = d >> 2, n = (d & 3) * 256 + ov;
      size_t idx = ((size_t)(b * 256 + vi)) * 1024 + n;
      v_arr[idx] = val;
      v_h[idx]   = (_Float16)val;
    }
  }
}

// =====================================================================
// K4: softmax over n for each (b,k)
// =====================================================================
__global__ __launch_bounds__(256)
void k_softmax_s(const float* __restrict__ P, _Float16* __restrict__ s_h)
{
  __shared__ float red[256];
  const int k = blockIdx.x, b = blockIdx.y, t = threadIdx.x;

  float vals[4];
  #pragma unroll
  for (int i = 0; i < 4; ++i) {
    int n = t + i * 256;
    int dl = n >> 4, op = n & 15;
    vals[i] = P[((size_t)(b * OTOT + 64 + op)) * DIMc + (k * 64 + dl)];
  }
  float m = fmaxf(fmaxf(vals[0], vals[1]), fmaxf(vals[2], vals[3]));
  red[t] = m; __syncthreads();
  for (int s = 128; s > 0; s >>= 1) {
    if (t < s) red[t] = fmaxf(red[t], red[t + s]);
    __syncthreads();
  }
  m = red[0]; __syncthreads();

  float e[4], sum = 0.f;
  #pragma unroll
  for (int i = 0; i < 4; ++i) { e[i] = __expf(vals[i] - m); sum += e[i]; }
  red[t] = sum; __syncthreads();
  for (int s = 128; s > 0; s >>= 1) {
    if (t < s) red[t] += red[t + s];
    __syncthreads();
  }
  const float inv = 1.f / red[0];
  #pragma unroll
  for (int i = 0; i < 4; ++i) {
    int n = t + i * 256;
    s_h[((size_t)(b * 16 + k)) * 1024 + n] = (_Float16)(e[i] * inv);
  }
}

// =====================================================================
// K5: content lambda  lam_c[b][k][v] = sum_n s[b,k,n] * v[b,v,n]  (WMMA)
// Fragments gathered with global_load_b128 (half-fragments contiguous).
// =====================================================================
__global__ __launch_bounds__(128)
void k_lamc(const _Float16* __restrict__ s_h, const _Float16* __restrict__ v_h,
            float* __restrict__ lam_c)
{
  const int b = blockIdx.y;
  const int w = threadIdx.x >> 5, lane = threadIdx.x & 31;
  const int vt = blockIdx.x * 4 + w;
  const int Ml = lane & 15, g = lane >> 4;

  const _Float16* sr = s_h + ((size_t)b * 16 + Ml) * 1024;                 // A row (k = Ml)
  const _Float16* vr = v_h + ((size_t)(b * 256 + vt * 16 + Ml)) * 1024;   // B col (v = Ml)

  v8f acc = {};
  for (int n0 = 0; n0 < 1024; n0 += 32) {
    v8h alo = *(const v8h*)(sr + n0 + 8 * g);
    v8h ahi = *(const v8h*)(sr + n0 + 16 + 8 * g);
    v16h af = CAT16(alo, ahi);
    v8h blo = *(const v8h*)(vr + n0 + 8 * g);
    v8h bhi = *(const v8h*)(vr + n0 + 16 + 8 * g);
    v16h bf = CAT16(blo, bhi);
    acc = __builtin_amdgcn_wmma_f32_16x16x32_f16(false, af, false, bf,
                                                 (short)0, acc, false, false);
  }
  #pragma unroll
  for (int r = 0; r < 8; ++r) {
    int k = r + 8 * g;
    lam_c[((size_t)(b * 16 + k)) * 256 + vt * 16 + Ml] = acc[r];
  }
}

// =====================================================================
// K6: positional conv as WMMA f32 16x16x4 + fused lambda apply
//  A (16k x 4taps) = emb chunk (lane-invariant, precomputed; 6 chunks, 23+1pad)
//  B (4taps x 16n) = shifted v-row window from LDS
//  C init = lam_c  ->  acc[r] = lam[k=r+8g][n=n0+Nl]
//  apply: q layout [b][h][n][k] -> each lane reads its 8 k's as 2x b128;
//  8 FMA/lane/head + shfl_xor(16) combines the two k-halves of the wave.
// =====================================================================
__global__ __launch_bounds__(128)
void k_out(const float* __restrict__ q_arr, const float* __restrict__ lam_c,
           const float* __restrict__ v_arr, const float* __restrict__ emb,
           float* __restrict__ out)
{
  __shared__ __align__(16) float vrow[1024 + 2 * PADc + 2];   // vrow[j] = v[j-11], zero-padded

  const int v = blockIdx.x, b = blockIdx.y, t = threadIdx.x;
  const int w = t >> 5, lane = t & 31;
  const int Nl = lane & 15, g = lane >> 4;

  const float* vb = v_arr + ((size_t)(b * 256 + v)) * 1024;
  for (int j = t; j < 1024 + 2 * PADc + 2; j += 128) {
    int src = j - PADc;
    vrow[j] = (src >= 0 && src < 1024) ? vb[src] : 0.f;
  }
  __syncthreads();

  // A fragments (emb): a[0] = A[M][mc+2g], a[1] = A[M][mc+2g+1]; M = k = Nl
  v2f aE[6];
  #pragma unroll
  for (int c = 0; c < 6; ++c) {
    int m0 = 4 * c + 2 * g;
    aE[c][0] = emb[Nl * MMc + m0];
    aE[c][1] = (m0 + 1 < MMc) ? emb[Nl * MMc + m0 + 1] : 0.f;  // tap 23 = zero pad
  }
  // lam_c feeds the accumulator directly
  float lam0[8];
  #pragma unroll
  for (int r = 0; r < 8; ++r)
    lam0[r] = lam_c[((size_t)(b * 16 + r + 8 * g)) * 256 + v];

  const float* qb = q_arr + (size_t)b * 4 * 1024 * 16;   // [h][n][k]
  float* ob = out + ((size_t)(b * 1024 + v)) * 1024;     // + h*256*1024 per head

  for (int tile = w * 16; tile < w * 16 + 16; ++tile) {
    const int n0 = tile * 16;
    v8f acc;
    #pragma unroll
    for (int r = 0; r < 8; ++r) acc[r] = lam0[r];

    #pragma unroll
    for (int c = 0; c < 6; ++c) {
      v2f bF;
      const int base = n0 + Nl + 4 * c + 2 * g;  // Vshift[m][n] = vrow[n+m]
      bF[0] = vrow[base];
      bF[1] = vrow[base + 1];
      acc = __builtin_amdgcn_wmma_f32_16x16x4_f32(false, aE[c], false, bF,
                                                  (short)0, acc, false, false);
    }

    float part[4];
    #pragma unroll
    for (int h = 0; h < 4; ++h) {
      // this lane's 8 k-values are contiguous: q[b][h][n0+Nl][8g .. 8g+7]
      const float* qp = qb + (((size_t)h * 1024) + n0 + Nl) * 16 + 8 * g;
      float4 qlo = *(const float4*)(qp);
      float4 qhi = *(const float4*)(qp + 4);
      float a = qlo.x * acc[0] + qlo.y * acc[1] + qlo.z * acc[2] + qlo.w * acc[3]
              + qhi.x * acc[4] + qhi.y * acc[5] + qhi.z * acc[6] + qhi.w * acc[7];
      part[h] = a;
    }
    #pragma unroll
    for (int h = 0; h < 4; ++h)
      part[h] += __shfl_xor(part[h], 16, 32);   // combine k=0..7 (g=0) with k=8..15 (g=1)

    if (g == 0) {
      #pragma unroll
      for (int h = 0; h < 4; ++h)
        ob[(size_t)h * 256 * 1024 + n0 + Nl] = part[h];
    }
  }
}

// =====================================================================
extern "C" void kernel_launch(void* const* d_in, const int* in_sizes, int n_in,
                              void* d_out, int out_size, void* d_ws, size_t ws_size,
                              hipStream_t stream) {
  (void)in_sizes; (void)n_in; (void)out_size; (void)ws_size;
  const float* x    = (const float*)d_in[0];
  const float* Wq   = (const float*)d_in[1];
  const float* bq   = (const float*)d_in[2];
  const float* gq   = (const float*)d_in[3];
  const float* betq = (const float*)d_in[4];
  const float* Wk   = (const float*)d_in[5];
  const float* bk   = (const float*)d_in[6];
  const float* Wv   = (const float*)d_in[7];
  const float* bv   = (const float*)d_in[8];
  const float* gv   = (const float*)d_in[9];
  const float* betv = (const float*)d_in[10];
  const float* emb  = (const float*)d_in[11];
  float* out = (float*)d_out;

  // workspace carve-up (float units); all region starts are 16B-aligned
  float* wsf   = (float*)d_ws;
  float* P     = wsf;                                   // 16*336*1024
  float* stats = P     + (size_t)16 * OTOT * DIMc;      // 4096
  float* q_arr = stats + 4096;                          // 16*4*1024*16  [b][h][n][k]
  float* v_arr = q_arr + (size_t)16 * 4 * 1024 * 16;    // 16*256*1024
  float* lam_c = v_arr + (size_t)16 * 256 * 1024;       // 16*16*256
  _Float16* v_h = (_Float16*)(lam_c + 16 * 16 * 256);   // 16*256*1024 halfs
  _Float16* s_h = v_h + (size_t)16 * 256 * 1024;        // 16*16*1024 halfs

  k_proj_gemm<<<dim3(DIMc / 64, (OTOT + 63) / 64, Bc), 128, 0, stream>>>(
      x, Wq, bq, Wk, bk, Wv, bv, P);
  k_bnstats<<<dim3(DIMc), 256, 0, stream>>>(P, stats);
  k_scatter_qv<<<dim3(DIMc, Bc), 256, 0, stream>>>(
      P, stats, gq, betq, gv, betv, q_arr, v_arr, v_h);
  k_softmax_s<<<dim3(KKc, Bc), 256, 0, stream>>>(P, s_h);
  k_lamc<<<dim3(4, Bc), 128, 0, stream>>>(s_h, v_h, lam_c);
  k_out<<<dim3(VVc, Bc), 128, 0, stream>>>(q_arr, lam_c, v_arr, emb, out);
}
